// EncoderDecoderLSTM_53970559041939
// MI455X (gfx1250) — compile-verified, matrix operand order checked by reading, not run on hardware
//
#include <hip/hip_runtime.h>
#include <stdint.h>

typedef __attribute__((ext_vector_type(16))) _Float16 v16h;
typedef __attribute__((ext_vector_type(8)))  _Float16 v8h;
typedef __attribute__((ext_vector_type(8)))  float    v8f;

#define Bsz  512
#define Hd   512
#define Tlen 256
#define HOR  64

union F16Frag { v16h v; v8h h[2]; };

__device__ __forceinline__ float sigmoidf_(float x) {
    return 1.0f / (1.0f + __expf(-x));
}
__device__ __forceinline__ float tanh_(float x) {
    // tanh(x) = 1 - 2/(exp(2x)+1); saturates correctly for |x| large
    return 1.0f - 2.0f / (__expf(2.0f * x) + 1.0f);
}

// ---------------------------------------------------------------------------
// One LSTM layer for one timestep:
//   gates(B,4H) = A1(B,H)@W1(4H,H)^T [+ A2(B,H)@W2(4H,H)^T] [+ x(B)*wih(4H)] + bias
//   i,f,g,o = act(gates); c = f*c + i*g; h_out = o*tanh(c)
// f16 operands, f32 accumulation via v_wmma_f32_16x16x32_f16.
// Each wave owns a 32(M batch) x 16(N hidden-unit) tile across all 4 gates:
// every B (weight) fragment feeds 2 WMMAs, halving L2 weight traffic vs M=16.
// ---------------------------------------------------------------------------
__global__ __launch_bounds__(256)
void lstm_step_kernel(const _Float16* __restrict__ A1, const _Float16* __restrict__ W1,
                      const _Float16* __restrict__ A2, const _Float16* __restrict__ W2,
                      const float* __restrict__ xptr, int x_stride,
                      const float* __restrict__ wih,     // (4H) scalar-input weights, may be null
                      const float* __restrict__ bias,    // (4H)
                      float* __restrict__ c,             // (B,H) f32, in-place
                      _Float16* __restrict__ h_out)      // (B,H) f16 (double-buffered by host)
{
    const int lane = threadIdx.x & 31;
    const int wave = threadIdx.x >> 5;
    const int tile = blockIdx.x * 8 + wave;   // 0..511
    const int b0 = (tile >> 5) * 32;          // batch tile base (32 rows)
    const int j0 = (tile & 31) * 16;          // hidden-unit tile base

    const int row = lane & 15;
    const int hi  = (lane >> 4) & 1;

    const v8f vzero = {0.f,0.f,0.f,0.f,0.f,0.f,0.f,0.f};
    v8f acc[2][4];
    #pragma unroll
    for (int hb = 0; hb < 2; ++hb)
        #pragma unroll
        for (int g = 0; g < 4; ++g) acc[hb][g] = vzero;

    const _Float16* Aarr[2] = {A1, A2};
    const _Float16* Warr[2] = {W1, W2};

    for (int m = 0; m < 2; ++m) {
        const _Float16* A = Aarr[m];
        const _Float16* W = Warr[m];
        if (A == nullptr) continue;
        const _Float16* arow0 = A + (size_t)(b0 + row) * Hd + hi * 8;
        const _Float16* arow1 = arow0 + (size_t)16 * Hd;
        #pragma unroll 2
        for (int kc = 0; kc < Hd; kc += 32) {
            F16Frag a0, a1;
            a0.h[0] = *(const v8h*)(arow0 + kc);
            a0.h[1] = *(const v8h*)(arow0 + kc + 16);
            a1.h[0] = *(const v8h*)(arow1 + kc);
            a1.h[1] = *(const v8h*)(arow1 + kc + 16);
            #pragma unroll
            for (int g = 0; g < 4; ++g) {
                const _Float16* wrow = W + (size_t)(g * Hd + j0 + row) * Hd + hi * 8 + kc;
                F16Frag bf;
                bf.h[0] = *(const v8h*)(wrow);
                bf.h[1] = *(const v8h*)(wrow + 16);
                acc[0][g] = __builtin_amdgcn_wmma_f32_16x16x32_f16(
                    false, a0.v, false, bf.v, (short)0, acc[0][g], false, false);
                acc[1][g] = __builtin_amdgcn_wmma_f32_16x16x32_f16(
                    false, a1.v, false, bf.v, (short)0, acc[1][g], false, false);
            }
        }
    }

    // Epilogue: per ISA C/D layout, acc[hb][g][r] is element
    //   (M = hb*16 + r + 8*hi, N = lane&15)
    const int n = lane & 15;
    const int j = j0 + n;
    const float bi = bias[j], bff = bias[j + Hd], bg = bias[j + 2*Hd], bo = bias[j + 3*Hd];
    float wi = 0.f, wf = 0.f, wg = 0.f, wo = 0.f;
    if (wih) { wi = wih[j]; wf = wih[j + Hd]; wg = wih[j + 2*Hd]; wo = wih[j + 3*Hd]; }

    #pragma unroll
    for (int hb = 0; hb < 2; ++hb) {
        #pragma unroll
        for (int r = 0; r < 8; ++r) {
            const int b = b0 + hb * 16 + r + hi * 8;
            const float xv = (xptr != nullptr) ? xptr[(size_t)b * x_stride] : 0.0f;
            float gi = acc[hb][0][r] + bi  + xv * wi;
            float gf = acc[hb][1][r] + bff + xv * wf;
            float gg = acc[hb][2][r] + bg  + xv * wg;
            float go = acc[hb][3][r] + bo  + xv * wo;
            gi = sigmoidf_(gi); gf = sigmoidf_(gf); gg = tanh_(gg); go = sigmoidf_(go);
            const size_t idx = (size_t)b * Hd + j;
            const float cn = gf * c[idx] + gi * gg;
            c[idx] = cn;
            h_out[idx] = (_Float16)(go * tanh_(cn));
        }
    }
}

// ---------------------------------------------------------------------------
// Head: out[b, s] = h1[b,:] . head_W + head_b ; also feeds decoder input.
// One wave per batch row (wave32 shuffle reduction).
// ---------------------------------------------------------------------------
__global__ __launch_bounds__(256)
void head_kernel(const _Float16* __restrict__ h1, const float* __restrict__ headW,
                 const float* __restrict__ headb, float* __restrict__ out,
                 float* __restrict__ dec_in, int s)
{
    const int lane = threadIdx.x & 31;
    const int b = blockIdx.x * 8 + (threadIdx.x >> 5);   // 64 blocks * 8 waves = 512
    const _Float16* hrow = h1 + (size_t)b * Hd + lane * 16;
    const float* wrow = headW + lane * 16;
    float sum = 0.f;
    #pragma unroll
    for (int k = 0; k < 16; ++k) sum += (float)hrow[k] * wrow[k];
    #pragma unroll
    for (int off = 16; off > 0; off >>= 1) sum += __shfl_xor(sum, off, 32);
    if (lane == 0) {
        const float v = sum + headb[0];
        out[(size_t)b * HOR + s] = v;
        dec_in[b] = v;
    }
}

__global__ __launch_bounds__(256)
void f32_to_f16_kernel(const float* __restrict__ in, _Float16* __restrict__ out, int n)
{
    const int i = blockIdx.x * 256 + threadIdx.x;
    if (i < n) out[i] = (_Float16)in[i];
}

__global__ __launch_bounds__(256)
void zero_kernel(uint32_t* __restrict__ p, int n)
{
    const int i = blockIdx.x * 256 + threadIdx.x;
    if (i < n) p[i] = 0u;
}

extern "C" void kernel_launch(void* const* d_in, const int* in_sizes, int n_in,
                              void* d_out, int out_size, void* d_ws, size_t ws_size,
                              hipStream_t stream) {
    (void)in_sizes; (void)n_in; (void)out_size; (void)ws_size;
    const float* x     = (const float*)d_in[0];
    const float* eWih0 = (const float*)d_in[1];
    const float* eWhh0 = (const float*)d_in[2];
    const float* eb0   = (const float*)d_in[3];
    const float* eWih1 = (const float*)d_in[4];
    const float* eWhh1 = (const float*)d_in[5];
    const float* eb1   = (const float*)d_in[6];
    const float* dWih0 = (const float*)d_in[7];
    const float* dWhh0 = (const float*)d_in[8];
    const float* db0   = (const float*)d_in[9];
    const float* dWih1 = (const float*)d_in[10];
    const float* dWhh1 = (const float*)d_in[11];
    const float* db1   = (const float*)d_in[12];
    const float* headW = (const float*)d_in[13];
    const float* headb = (const float*)d_in[14];
    float* out = (float*)d_out;

    // --- workspace layout (~16 MB) ---
    const size_t Welem = (size_t)4 * Hd * Hd;   // 2048*512 elements per weight matrix
    char* ws = (char*)d_ws;
    _Float16* w16 = (_Float16*)ws;
    _Float16* whh0e = w16 + 0 * Welem;
    _Float16* wih1e = w16 + 1 * Welem;
    _Float16* whh1e = w16 + 2 * Welem;
    _Float16* whh0d = w16 + 3 * Welem;
    _Float16* wih1d = w16 + 4 * Welem;
    _Float16* whh1d = w16 + 5 * Welem;
    char* p = ws + 6 * Welem * sizeof(_Float16);
    float* c0 = (float*)p;            p += (size_t)Bsz * Hd * sizeof(float);
    float* c1 = (float*)p;            p += (size_t)Bsz * Hd * sizeof(float);
    _Float16* h0buf[2];
    h0buf[0] = (_Float16*)p;          p += (size_t)Bsz * Hd * sizeof(_Float16);
    h0buf[1] = (_Float16*)p;          p += (size_t)Bsz * Hd * sizeof(_Float16);
    _Float16* h1buf[2];
    h1buf[0] = (_Float16*)p;          p += (size_t)Bsz * Hd * sizeof(_Float16);
    h1buf[1] = (_Float16*)p;          p += (size_t)Bsz * Hd * sizeof(_Float16);
    float* dec_in = (float*)p;

    // --- convert recurrent/input weight matrices to f16 ---
    const int cgrid = (int)(Welem / 256);
    f32_to_f16_kernel<<<cgrid, 256, 0, stream>>>(eWhh0, whh0e, (int)Welem);
    f32_to_f16_kernel<<<cgrid, 256, 0, stream>>>(eWih1, wih1e, (int)Welem);
    f32_to_f16_kernel<<<cgrid, 256, 0, stream>>>(eWhh1, whh1e, (int)Welem);
    f32_to_f16_kernel<<<cgrid, 256, 0, stream>>>(dWhh0, whh0d, (int)Welem);
    f32_to_f16_kernel<<<cgrid, 256, 0, stream>>>(dWih1, wih1d, (int)Welem);
    f32_to_f16_kernel<<<cgrid, 256, 0, stream>>>(dWhh1, whh1d, (int)Welem);

    // --- zero c0,c1 and the h ping-pong buffers (4 MB contiguous) ---
    const int zn = (int)((2 * (size_t)Bsz * Hd * sizeof(float)
                        + 4 * (size_t)Bsz * Hd * sizeof(_Float16)) / 4);
    zero_kernel<<<(zn + 255) / 256, 256, 0, stream>>>((uint32_t*)c0, zn);

    int cur0 = 0, cur1 = 0;

    // --- encoder: 256 timesteps, 2 layers per step ---
    for (int t = 0; t < Tlen; ++t) {
        lstm_step_kernel<<<64, 256, 0, stream>>>(
            h0buf[cur0], whh0e, nullptr, nullptr,
            x + t, Tlen, eWih0, eb0, c0, h0buf[cur0 ^ 1]);
        lstm_step_kernel<<<64, 256, 0, stream>>>(
            h1buf[cur1], whh1e, h0buf[cur0 ^ 1], wih1e,
            nullptr, 0, nullptr, eb1, c1, h1buf[cur1 ^ 1]);
        cur0 ^= 1; cur1 ^= 1;
    }

    // --- decoder: 64 autoregressive steps ---
    for (int s = 0; s < HOR; ++s) {
        const float* xp = (s == 0) ? (x + (Tlen - 1)) : dec_in;
        const int xs = (s == 0) ? Tlen : 1;
        lstm_step_kernel<<<64, 256, 0, stream>>>(
            h0buf[cur0], whh0d, nullptr, nullptr,
            xp, xs, dWih0, db0, c0, h0buf[cur0 ^ 1]);
        lstm_step_kernel<<<64, 256, 0, stream>>>(
            h1buf[cur1], whh1d, h0buf[cur0 ^ 1], wih1d,
            nullptr, 0, nullptr, db1, c1, h1buf[cur1 ^ 1]);
        cur0 ^= 1; cur1 ^= 1;
        head_kernel<<<64, 256, 0, stream>>>(h1buf[cur1], headW, headb, out, dec_in, s);
    }
}
